// AttnBlock_2104533975764
// MI455X (gfx1250) — matrix operand order repaired and checked by reading, not compiled, not run on hardware
//
#include <hip/hip_runtime.h>
#include <math.h>

// ---------------------------------------------------------------------------
// MI455X (gfx1250) ViT encoder, f16 WMMA (f32 accum) everywhere.
//   t (B,Npad,C) f32 persistent tokens; Npad=1760 (pad of N=1729 to x32).
//   Weights pre-transposed to f16 (N x K) so every WMMA B-fragment is two
//   contiguous 16B loads; activations row-major (M x K) so every A-fragment
//   is two contiguous 16B loads (per CDNA5 16-bit A/B VGPR layouts).
//   V is written transposed (hd x N) directly by the QKV GEMM epilogue.
//   Attention: flash-style; q pre-scaled by 1/sqrt(hd); per-lane deferred
//   softmax denominator (one cross-lane reduction at the end); the masked
//   tail chunk is peeled out of the main loop (no masking VALU in 54/55
//   iterations).
// ---------------------------------------------------------------------------

typedef _Float16 half_t;
typedef __attribute__((ext_vector_type(16))) _Float16 v16h;
typedef __attribute__((ext_vector_type(8)))  _Float16 v8h;
typedef __attribute__((ext_vector_type(8)))  float    v8f;

__device__ __forceinline__ v8f wmma_f16(v16h a, v16h b, v8f c) {
  return __builtin_amdgcn_wmma_f32_16x16x32_f16(
      /*neg_a=*/false, a, /*neg_b=*/false, b,
      /*c_mod=*/(short)0, c, /*reuse_a=*/false, /*reuse_b=*/false);
}

__device__ __forceinline__ v16h combine16(v8h lo, v8h hi) {
  v16h r;
#pragma unroll
  for (int i = 0; i < 8; ++i) { r[i] = lo[i]; r[i + 8] = hi[i]; }
  return r;
}

// A-fragment (16x32 f16, M x K), A row-major with leading dim lda (halfs).
// Lane L: row = m0 + L%16; halves K = [8h..8h+7] and [16+8h..16+8h+7], h=L/16.
__device__ __forceinline__ v16h load_a(const half_t* A, int lda, int m0, int k0,
                                       int lane) {
  const int row = m0 + (lane & 15);
  const int hs  = lane >> 4;
  const half_t* p = A + (size_t)row * lda + k0 + 8 * hs;
  return combine16(*(const v8h*)p, *(const v8h*)(p + 16));
}

// B-fragment (32x16 f16, K x N) sourced from transposed storage Bt (N x K).
// Lane L: col = n0 + L%16; K = [k0+8h..k0+8h+7] and [k0+16+8h..k0+16+8h+7], h=L/16.
__device__ __forceinline__ v16h load_bt(const half_t* Bt, int ldb, int n0,
                                        int k0, int lane) {
  const int col = n0 + (lane & 15);
  const int hs  = lane >> 4;
  const half_t* p = Bt + (size_t)col * ldb + k0 + 8 * hs;
  return combine16(*(const v8h*)p, *(const v8h*)(p + 16));
}

// ---------------------------------------------------------------------------
// Weight transpose + f32->f16 convert:  W[D][K][N] -> Wt[D][N][K]
// ---------------------------------------------------------------------------
__global__ void transpose_cvt_kernel(const float* __restrict__ W,
                                     half_t* __restrict__ Wt, int K, int Nn) {
  size_t idx = (size_t)blockIdx.x * 256 + threadIdx.x;
  int k = (int)(idx % K);
  size_t r = idx / K;
  int n = (int)(r % Nn);
  int d = (int)(r / Nn);
  Wt[idx] = (half_t)W[((size_t)d * K + k) * Nn + n];
}

// ---------------------------------------------------------------------------
// Patch embed: t[b][0][c]=cls; t[b][1+s][c] = sum_k x[b][k][s]*Wpe[k][c]+bpe[c]
// pad rows -> 0.  (K=64; scalar f32, trivial fraction of total work)
// ---------------------------------------------------------------------------
__global__ void patch_embed_kernel(const float* __restrict__ x,
                                   const float* __restrict__ Wpe,
                                   const float* __restrict__ bpe,
                                   const float* __restrict__ cls,
                                   float* __restrict__ t, int Npad) {
  int idx = blockIdx.x * 256 + threadIdx.x;  // over B*Npad*256
  int c = idx & 255;
  int row = idx >> 8;              // b*Npad + n
  int n = row % Npad, b = row / Npad;
  float v;
  if (n == 0) {
    v = cls[c];
  } else if (n <= 1728) {
    int s = n - 1;
    float acc = bpe[c];
    const float* xb = x + (size_t)b * 64 * 1728 + s;
#pragma unroll 8
    for (int k = 0; k < 64; ++k) acc += xb[(size_t)k * 1728] * Wpe[k * 256 + c];
    v = acc;
  } else {
    v = 0.0f;
  }
  t[(size_t)row * 256 + c] = v;
}

// ---------------------------------------------------------------------------
// LayerNorm over C=256 -> f16; pad rows zeroed (keeps WMMA inputs finite).
// One 256-thread block per token row.
// ---------------------------------------------------------------------------
__global__ void ln_kernel(const float* __restrict__ t,
                          const float* __restrict__ g,
                          const float* __restrict__ bb,
                          half_t* __restrict__ y, int Npad, int Nreal) {
  const int row = blockIdx.x;
  const int n = row % Npad;
  const int tid = threadIdx.x;
  if (n >= Nreal) { y[(size_t)row * 256 + tid] = (half_t)0.0f; return; }
  float v = t[(size_t)row * 256 + tid];
  float s = v, q = v * v;
#pragma unroll
  for (int m = 16; m; m >>= 1) { s += __shfl_xor(s, m); q += __shfl_xor(q, m); }
  __shared__ float rs[8], rq[8];
  int wave = tid >> 5, lane = tid & 31;
  if (lane == 0) { rs[wave] = s; rq[wave] = q; }
  __syncthreads();
  float S = 0.0f, Q = 0.0f;
#pragma unroll
  for (int i = 0; i < 8; ++i) { S += rs[i]; Q += rq[i]; }
  float mean = S * (1.0f / 256.0f);
  float var  = Q * (1.0f / 256.0f) - mean * mean;
  float inv  = rsqrtf(var + 1e-5f);
  y[(size_t)row * 256 + tid] = (half_t)((v - mean) * inv * g[tid] + bb[tid]);
}

// ---------------------------------------------------------------------------
// Generic WMMA GEMM:  out[m][n] = sum_k A[m][k]*Bt[n][k] (+bias) (+epilogue)
//  MODE 1: t_f32[m][n] += acc + bias[n]   (residual)
//  MODE 2: store f16 gelu(acc + bias[n])  (FFN hidden, exact erf gelu)
//  MODE 3: QKV: cols [0,512) -> outH (stride 768); cols [512,768) -> vT
//          (V stored transposed per head as (hd x Npad)); no bias.
// Block: 256 threads / 8 waves; wave w -> rows [w*16,w*16+16), 64 cols/block.
// ---------------------------------------------------------------------------
template <int MODE>
__global__ void __launch_bounds__(256)
gemm_kernel(const half_t* __restrict__ A, const half_t* __restrict__ Bt,
            const float* __restrict__ bias, half_t* __restrict__ outH,
            float* __restrict__ outF, half_t* __restrict__ vTout,
            int K, int Nout, int NpadP) {
  const int lane = threadIdx.x & 31;
  const int wave = threadIdx.x >> 5;
  const int m0 = blockIdx.x * 128 + wave * 16;
  const int n_base = blockIdx.y * 64;
  const int colL = lane & 15, hs = lane >> 4;
  v8f acc[4] = {};
  for (int k0 = 0; k0 < K; k0 += 32) {
    if (k0 + 32 < K) {  // prefetch next K slice (global_prefetch_b8)
      __builtin_prefetch(A + (size_t)(m0 + colL) * K + k0 + 32 + 8 * hs, 0, 3);
      __builtin_prefetch(Bt + (size_t)(n_base + colL) * K + k0 + 32 + 16 * hs, 0, 3);
    }
    v16h a = load_a(A, K, m0, k0, lane);
#pragma unroll
    for (int t4 = 0; t4 < 4; ++t4) {
      v16h b = load_bt(Bt, K, n_base + 16 * t4, k0, lane);
      acc[t4] = wmma_f16(a, b, acc[t4]);
    }
  }
#pragma unroll
  for (int t4 = 0; t4 < 4; ++t4) {
    const int ncol0 = n_base + 16 * t4;
    const int n = ncol0 + colL;
    if (MODE == 3) {
      if (ncol0 >= 512) {  // V block -> transposed store (uniform per tile)
        const int bidx = m0 / NpadP;
        const int tokb = m0 - bidx * NpadP;
        const int cv = n - 512;
        const int hh = cv >> 5, dd = cv & 31;
        half_t* vp = vTout + ((size_t)(bidx * 8 + hh) * 32 + dd) * NpadP + tokb;
#pragma unroll
        for (int r = 0; r < 8; ++r) vp[r + 8 * hs] = (half_t)acc[t4][r];
      } else {             // Q/K block -> qkv buffer
#pragma unroll
        for (int r = 0; r < 8; ++r) {
          int m = m0 + r + 8 * hs;
          outH[(size_t)m * Nout + n] = (half_t)acc[t4][r];
        }
      }
    } else {
      float bv = bias[n];
#pragma unroll
      for (int r = 0; r < 8; ++r) {
        int m = m0 + r + 8 * hs;
        float v = acc[t4][r] + bv;
        if (MODE == 1) {
          outF[(size_t)m * Nout + n] += v;
        } else {
          float gel = 0.5f * v * (1.0f + erff(v * 0.70710678118654752f));
          outH[(size_t)m * Nout + n] = (half_t)gel;
        }
      }
    }
  }
}

// ---------------------------------------------------------------------------
// Flash attention, one wave per (b, head, 16-row q tile).
// Scores: q(16xhd32) x K^T via v_wmma_f32_16x16x32_f16 (K-dim == hd == 32).
// q pre-scaled by 1/sqrt(hd) in the A-fragment. Row max reduced per chunk
// (butterfly over 16-lane halves); softmax denominator kept per-lane (alpha
// is row-uniform, recurrence is linear) and reduced once at the end.
// P goes through 1KB LDS to convert C-layout -> A-layout for the P.V WMMAs.
// The masked tail chunk is peeled: the main loop has no masking code at all.
// ---------------------------------------------------------------------------
__global__ void __launch_bounds__(32)
attn_kernel(const half_t* __restrict__ qkv, const half_t* __restrict__ vT,
            half_t* __restrict__ o, int Npad, int Nreal) {
  __shared__ alignas(16) half_t Plds[16 * 32];
  const int lane = threadIdx.x;
  const int bh = blockIdx.x;
  const int b = bh >> 3, h = bh & 7;
  const int m0 = blockIdx.y * 16;
  const half_t* qk = qkv + (size_t)b * Npad * 768;
  const half_t* Qb = qk + h * 32;          // lda = 768
  const half_t* Kb = qk + 256 + h * 32;    // transposed-B view, ldb = 768
  const half_t* Vt = vT + (size_t)bh * 32 * Npad;  // (hd x Npad), ldb = Npad

  v16h qa = load_a(Qb, 768, m0, 0, lane);
  qa = qa * (_Float16)0.17677669529663687f;  // fold 1/sqrt(hd) into q
  v8f acc0 = {}, acc1 = {};
  float mr[8], lr[8];
#pragma unroll
  for (int r = 0; r < 8; ++r) { mr[r] = -3.0e38f; lr[r] = 0.0f; }
  const int colL = lane & 15, hs = lane >> 4;
  const int Nfull = Nreal & ~31;  // 1728: last full unmasked chunk boundary

  int key0 = 0;
  for (; key0 < Nfull; key0 += 32) {  // ---- main loop: no masking ----
    v16h kb0 = load_bt(Kb, 768, key0, 0, lane);
    v16h kb1 = load_bt(Kb, 768, key0 + 16, 0, lane);
    if (key0 + 32 < Npad) {  // prefetch next chunk's K / V lines
      const half_t* pk = Kb + (size_t)(key0 + 32 + colL) * 768 + 16 * hs;
      __builtin_prefetch(pk, 0, 3);
      __builtin_prefetch(pk + (size_t)16 * 768, 0, 3);
      const half_t* pv = Vt + (size_t)colL * Npad + key0 + 32 + 16 * hs;
      __builtin_prefetch(pv, 0, 3);
      __builtin_prefetch(pv + (size_t)16 * Npad, 0, 3);
    }
    v8f z0 = {}, z1 = {};
    v8f s0 = wmma_f16(qa, kb0, z0);
    v8f s1 = wmma_f16(qa, kb1, z1);
#pragma unroll
    for (int r = 0; r < 8; ++r) {
      float a0 = s0[r];
      float a1 = s1[r];
      float mx = fmaxf(a0, a1);
      mx = fmaxf(mx, __shfl_xor(mx, 1));
      mx = fmaxf(mx, __shfl_xor(mx, 2));
      mx = fmaxf(mx, __shfl_xor(mx, 4));
      mx = fmaxf(mx, __shfl_xor(mx, 8));
      float mnew  = fmaxf(mr[r], mx);
      float alpha = __expf(mr[r] - mnew);
      mr[r] = mnew;
      float p0 = __expf(a0 - mnew);
      float p1 = __expf(a1 - mnew);
      lr[r] = lr[r] * alpha + (p0 + p1);   // per-lane partial denominator
      acc0[r] *= alpha;
      acc1[r] *= alpha;
      int prow = r + 8 * hs;
      Plds[prow * 32 + colL]      = (half_t)p0;
      Plds[prow * 32 + 16 + colL] = (half_t)p1;
    }
    __syncthreads();
    v16h pa  = load_a(Plds, 32, 0, 0, lane);
    v16h vb0 = load_bt(Vt, Npad, 0, key0, lane);
    v16h vb1 = load_bt(Vt, Npad, 16, key0, lane);
    acc0 = wmma_f16(pa, vb0, acc0);
    acc1 = wmma_f16(pa, vb1, acc1);
    __syncthreads();
  }

  for (; key0 < Npad; key0 += 32) {  // ---- peeled masked tail chunk(s) ----
    v16h kb0 = load_bt(Kb, 768, key0, 0, lane);
    v16h kb1 = load_bt(Kb, 768, key0 + 16, 0, lane);
    v8f z0 = {}, z1 = {};
    v8f s0 = wmma_f16(qa, kb0, z0);
    v8f s1 = wmma_f16(qa, kb1, z1);
    const bool m0k = (key0 + colL) >= Nreal;
    const bool m1k = (key0 + 16 + colL) >= Nreal;
#pragma unroll
    for (int r = 0; r < 8; ++r) {
      float a0 = m0k ? -1.0e30f : s0[r];
      float a1 = m1k ? -1.0e30f : s1[r];
      float mx = fmaxf(a0, a1);
      mx = fmaxf(mx, __shfl_xor(mx, 1));
      mx = fmaxf(mx, __shfl_xor(mx, 2));
      mx = fmaxf(mx, __shfl_xor(mx, 4));
      mx = fmaxf(mx, __shfl_xor(mx, 8));
      float mnew  = fmaxf(mr[r], mx);
      float alpha = __expf(mr[r] - mnew);
      mr[r] = mnew;
      float p0 = __expf(a0 - mnew);
      float p1 = __expf(a1 - mnew);
      lr[r] = lr[r] * alpha + (p0 + p1);
      acc0[r] *= alpha;
      acc1[r] *= alpha;
      int prow = r + 8 * hs;
      Plds[prow * 32 + colL]      = (half_t)p0;
      Plds[prow * 32 + 16 + colL] = (half_t)p1;
    }
    __syncthreads();
    v16h pa  = load_a(Plds, 32, 0, 0, lane);
    v16h vb0 = load_bt(Vt, Npad, 0, key0, lane);
    v16h vb1 = load_bt(Vt, Npad, 16, key0, lane);
    acc0 = wmma_f16(pa, vb0, acc0);
    acc1 = wmma_f16(pa, vb1, acc1);
    __syncthreads();
  }

  half_t* ob = o + ((size_t)b * Npad + m0) * 256 + h * 32;
#pragma unroll
  for (int r = 0; r < 8; ++r) {
    int row = r + 8 * hs;
    float l = lr[r];   // reduce the deferred denominator once
    l += __shfl_xor(l, 1);
    l += __shfl_xor(l, 2);
    l += __shfl_xor(l, 4);
    l += __shfl_xor(l, 8);
    float inv = 1.0f / l;
    ob[(size_t)row * 256 + colL]      = (half_t)(acc0[r] * inv);
    ob[(size_t)row * 256 + 16 + colL] = (half_t)(acc1[r] * inv);
  }
}

// ---------------------------------------------------------------------------
// Final LN + scatter: row 0 -> x_cls[b][c]; rows 1..1728 -> feat[b][c][s].
// ---------------------------------------------------------------------------
__global__ void final_kernel(const float* __restrict__ t,
                             const float* __restrict__ g,
                             const float* __restrict__ bb,
                             float* __restrict__ out, int Npad, int Nreal) {
  const int row = blockIdx.x;
  const int bidx = row / Npad;
  const int n = row % Npad;
  if (n >= Nreal) return;
  const int tid = threadIdx.x;
  float v = t[(size_t)row * 256 + tid];
  float s = v, q = v * v;
#pragma unroll
  for (int m = 16; m; m >>= 1) { s += __shfl_xor(s, m); q += __shfl_xor(q, m); }
  __shared__ float rs[8], rq[8];
  int wave = tid >> 5, lane = tid & 31;
  if (lane == 0) { rs[wave] = s; rq[wave] = q; }
  __syncthreads();
  float S = 0.0f, Q = 0.0f;
#pragma unroll
  for (int i = 0; i < 8; ++i) { S += rs[i]; Q += rq[i]; }
  float mean = S * (1.0f / 256.0f);
  float var  = Q * (1.0f / 256.0f) - mean * mean;
  float inv  = rsqrtf(var + 1e-5f);
  float val = (v - mean) * inv * g[tid] + bb[tid];
  if (n == 0) out[(size_t)bidx * 256 + tid] = val;
  else out[1024 + ((size_t)bidx * 256 + tid) * 1728 + (n - 1)] = val;
}

// ---------------------------------------------------------------------------
extern "C" void kernel_launch(void* const* d_in, const int* in_sizes, int n_in,
                              void* d_out, int out_size, void* d_ws,
                              size_t ws_size, hipStream_t stream) {
  const float* x     = (const float*)d_in[0];
  const float* Wpe   = (const float*)d_in[1];
  const float* bpe   = (const float*)d_in[2];
  const float* cls   = (const float*)d_in[3];
  const float* ln1g  = (const float*)d_in[4];
  const float* ln1b  = (const float*)d_in[5];
  const float* Wqkv  = (const float*)d_in[6];
  const float* Wproj = (const float*)d_in[7];
  const float* bproj = (const float*)d_in[8];
  const float* ln2g  = (const float*)d_in[9];
  const float* ln2b  = (const float*)d_in[10];
  const float* W1    = (const float*)d_in[11];
  const float* b1    = (const float*)d_in[12];
  const float* W2    = (const float*)d_in[13];
  const float* b2    = (const float*)d_in[14];
  const float* nfg   = (const float*)d_in[15];
  const float* nfb   = (const float*)d_in[16];
  float* out = (float*)d_out;

  constexpr int B = 4, C = 256, Npad = 1760, Nreal = 1729, D = 2;
  const int Mtot = B * Npad;  // 7040

  char* ws = (char*)d_ws;
  size_t off = 0;
  auto carve = [&](size_t bytes) {
    char* p = ws + off;
    off = (off + bytes + 255) & ~(size_t)255;
    return p;
  };
  float*  t      = (float*) carve((size_t)Mtot * C * 4);
  half_t* y      = (half_t*)carve((size_t)Mtot * C * 2);
  half_t* hbuf   = (half_t*)carve((size_t)Mtot * C * 2);
  half_t* obuf   = (half_t*)carve((size_t)Mtot * C * 2);
  half_t* qkvb   = (half_t*)carve((size_t)Mtot * 3 * C * 2);
  half_t* vT     = (half_t*)carve((size_t)B * 8 * 32 * Npad * 2);
  half_t* WqkvT  = (half_t*)carve((size_t)D * 768 * 256 * 2);
  half_t* WprojT = (half_t*)carve((size_t)D * 256 * 256 * 2);
  half_t* W1T    = (half_t*)carve((size_t)D * 256 * 256 * 2);
  half_t* W2T    = (half_t*)carve((size_t)D * 256 * 256 * 2);

  transpose_cvt_kernel<<<D * 256 * 768 / 256, 256, 0, stream>>>(Wqkv, WqkvT, 256, 768);
  transpose_cvt_kernel<<<D * 256 * 256 / 256, 256, 0, stream>>>(Wproj, WprojT, 256, 256);
  transpose_cvt_kernel<<<D * 256 * 256 / 256, 256, 0, stream>>>(W1, W1T, 256, 256);
  transpose_cvt_kernel<<<D * 256 * 256 / 256, 256, 0, stream>>>(W2, W2T, 256, 256);

  patch_embed_kernel<<<Mtot, 256, 0, stream>>>(x, Wpe, bpe, cls, t, Npad);

  for (int d = 0; d < D; ++d) {
    ln_kernel<<<Mtot, 256, 0, stream>>>(t, ln1g + d * C, ln1b + d * C, y, Npad, Nreal);
    gemm_kernel<3><<<dim3(Mtot / 128, 768 / 64), 256, 0, stream>>>(
        y, WqkvT + (size_t)d * 768 * 256, nullptr, qkvb, nullptr, vT, 256, 768, Npad);
    attn_kernel<<<dim3(B * 8, Npad / 16), 32, 0, stream>>>(qkvb, vT, obuf, Npad, Nreal);
    gemm_kernel<1><<<dim3(Mtot / 128, 4), 256, 0, stream>>>(
        obuf, WprojT + (size_t)d * 65536, bproj + d * C, nullptr, t, nullptr, 256, 256, Npad);
    ln_kernel<<<Mtot, 256, 0, stream>>>(t, ln2g + d * C, ln2b + d * C, y, Npad, Nreal);
    gemm_kernel<2><<<dim3(Mtot / 128, 4), 256, 0, stream>>>(
        y, W1T + (size_t)d * 65536, b1 + d * C, hbuf, nullptr, nullptr, 256, 256, Npad);
    gemm_kernel<1><<<dim3(Mtot / 128, 4), 256, 0, stream>>>(
        hbuf, W2T + (size_t)d * 65536, b2 + d * C, nullptr, t, nullptr, 256, 256, Npad);
  }

  final_kernel<<<Mtot, 256, 0, stream>>>(t, nfg, nfb, out, Npad, Nreal);
}
